// GraphConvSparse_23510650978717
// MI455X (gfx1250) — matrix operand order repaired and checked by reading, not compiled, not run on hardware
//
#include <hip/hip_runtime.h>

#define D_IN  512
#define D_OUT 64
#define KPANEL 128
#define KPAD  (KPANEL + 8)   // +16B pad: 272B lane stride -> conflict-free banks, keeps 16B alignment

typedef __attribute__((ext_vector_type(16))) __bf16 v16bf;
typedef __attribute__((ext_vector_type(8)))  __bf16 v8bf;
typedef __attribute__((ext_vector_type(8)))  float  v8f;

static __device__ __forceinline__ v16bf lds_b_frag(const __bf16* p) {
  // two 16B (alignment-safe) LDS vector loads -> one 32B fragment
  v8bf x0 = *(const v8bf*)p;
  v8bf x1 = *(const v8bf*)(p + 16);
  v16bf r;
  #pragma unroll
  for (int j = 0; j < 8; ++j) { r[j] = x0[j]; r[8 + j] = x1[j]; }
  return r;
}

static __device__ __forceinline__ void a_frag_split(const float* arow, int koff,
                                                    v16bf& hi, v16bf& lo) {
  // 16-bit A 16x32 layout: lane<16 holds K 0-7 & 16-23, lane>=16 holds K 8-15 & 24-31
  const float4* a4 = (const float4*)(arow + koff);
  float4 q0 = a4[0], q1 = a4[1], q2 = a4[4], q3 = a4[5];
  float av[16] = { q0.x,q0.y,q0.z,q0.w, q1.x,q1.y,q1.z,q1.w,
                   q2.x,q2.y,q2.z,q2.w, q3.x,q3.y,q3.z,q3.w };
  #pragma unroll
  for (int j = 0; j < 16; ++j) {
    __bf16 h = (__bf16)av[j];            // hardware RNE cvt
    hi[j] = h;
    lo[j] = (__bf16)(av[j] - (float)h);  // residual term
  }
}

#define WMMA_BF16(ACC, AF, BF) \
  ACC = __builtin_amdgcn_wmma_f32_16x16x32_bf16(false, AF, false, BF, (short)0, ACC, false, false)

// ---------------------------------------------------------------------------
// h = inputs @ weight, fp32-accurate split-bf16: A*B ~= Ahi*Bhi + Ahi*Blo + Alo*Bhi
// One wave owns a 32-row strip (2 M-tiles) x 64 cols (4 N-tiles): every B
// fragment feeds 6 WMMAs; 24 v_wmma per 32-wide K step.
// ---------------------------------------------------------------------------
__global__ __launch_bounds__(256) void gemm_bf16x3(const float* __restrict__ A,
                                                   const float* __restrict__ W,
                                                   float* __restrict__ H, int n) {
  __shared__ __bf16 sBhi[D_OUT * KPAD];
  __shared__ __bf16 sBlo[D_OUT * KPAD];

  const int lane = threadIdx.x & 31;
  const int wave = threadIdx.x >> 5;     // 0..7
  const int hsel = lane >> 4;            // K half select
  const int l15  = lane & 15;
  const int m0   = (blockIdx.x * 8 + wave) * 32;

  v8f acc0[4] = {}, acc1[4] = {};

  int r0 = m0 + l15;      if (r0 >= n) r0 = n - 1;   // clamp loads; stores guarded
  int r1 = m0 + 16 + l15; if (r1 >= n) r1 = n - 1;
  const float* arow0 = A + (size_t)r0 * D_IN;
  const float* arow1 = A + (size_t)r1 * D_IN;

  for (int kp = 0; kp < D_IN; kp += KPANEL) {
    // stage weight K-panel into LDS as split bf16, transposed [n][k] (padded)
    for (int i = threadIdx.x; i < KPANEL * D_OUT; i += 256) {
      int k  = i >> 6;                   // panel-local K
      int nn = i & 63;                   // output column
      float w = W[(size_t)kp * D_OUT + i];
      __bf16 hi = (__bf16)w;
      sBhi[nn * KPAD + k] = hi;
      sBlo[nn * KPAD + k] = (__bf16)(w - (float)hi);
    }
    __syncthreads();

    if (m0 < n) {
      for (int ks = 0; ks < KPANEL; ks += 32) {
        v16bf a0hi, a0lo, a1hi, a1lo;
        a_frag_split(arow0, kp + ks + hsel * 8, a0hi, a0lo);
        a_frag_split(arow1, kp + ks + hsel * 8, a1hi, a1lo);

        const __bf16* bh = &sBhi[l15 * KPAD + ks + hsel * 8];
        const __bf16* bl = &sBlo[l15 * KPAD + ks + hsel * 8];
        #pragma unroll
        for (int t = 0; t < 4; ++t) {
          v16bf bhi = lds_b_frag(bh + t * 16 * KPAD);
          v16bf blo = lds_b_frag(bl + t * 16 * KPAD);
          WMMA_BF16(acc0[t], a0hi, bhi);
          WMMA_BF16(acc0[t], a0hi, blo);
          WMMA_BF16(acc0[t], a0lo, bhi);
          WMMA_BF16(acc1[t], a1hi, bhi);
          WMMA_BF16(acc1[t], a1hi, blo);
          WMMA_BF16(acc1[t], a1lo, bhi);
        }
      }
    }
    __syncthreads();
  }

  // C/D layout: VGPR j -> M = j (lanes 0-15) / j+8 (lanes 16-31), N = l15
  if (m0 + 32 <= n) {                    // fast path: no per-row guards
    #pragma unroll
    for (int t = 0; t < 4; ++t) {
      #pragma unroll
      for (int j = 0; j < 8; ++j) {
        int row = m0 + j + hsel * 8;
        H[(size_t)row * D_OUT + t * 16 + l15]        = acc0[t][j];
        H[(size_t)(row + 16) * D_OUT + t * 16 + l15] = acc1[t][j];
      }
    }
  } else if (m0 < n) {                   // tail strip
    #pragma unroll
    for (int t = 0; t < 4; ++t) {
      #pragma unroll
      for (int j = 0; j < 8; ++j) {
        int row = m0 + j + hsel * 8;
        if (row < n)      H[(size_t)row * D_OUT + t * 16 + l15]        = acc0[t][j];
        if (row + 16 < n) H[(size_t)(row + 16) * D_OUT + t * 16 + l15] = acc1[t][j];
      }
    }
  }
}

// ---------------------------------------------------------------------------
// COO scatter: one wave per edge; edge metadata scalarized (SMEM loads);
// each lane handles 2 channels. h gathers hit the 192MB L2; f32 L2 atomics.
// ---------------------------------------------------------------------------
__global__ __launch_bounds__(256) void spmm_scatter(const float* __restrict__ H,
                                                    const float* __restrict__ vals,
                                                    const int* __restrict__ rows,
                                                    const int* __restrict__ cols,
                                                    float* __restrict__ out, int E) {
  int tid  = blockIdx.x * blockDim.x + threadIdx.x;
  int e    = __builtin_amdgcn_readfirstlane(tid >> 5);   // wave-uniform edge id
  int lane = threadIdx.x & 31;
  if (e >= E) return;
  float v = vals[e];
  int   r = rows[e];
  int   c = cols[e];
  const float2 h2 = *(const float2*)(H + (size_t)c * D_OUT + lane * 2);
  float* o = out + (size_t)r * D_OUT + lane * 2;
  __hip_atomic_fetch_add(o,     v * h2.x, __ATOMIC_RELAXED, __HIP_MEMORY_SCOPE_AGENT);
  __hip_atomic_fetch_add(o + 1, v * h2.y, __ATOMIC_RELAXED, __HIP_MEMORY_SCOPE_AGENT);
}

__global__ __launch_bounds__(256) void zero4_f32(float4* __restrict__ p, int n4) {
  int i = blockIdx.x * blockDim.x + threadIdx.x;
  if (i < n4) p[i] = make_float4(0.f, 0.f, 0.f, 0.f);
}

__global__ __launch_bounds__(256) void relu4_f32(float4* __restrict__ p, int n4) {
  int i = blockIdx.x * blockDim.x + threadIdx.x;
  if (i < n4) {
    float4 v = p[i];
    v.x = fmaxf(v.x, 0.f); v.y = fmaxf(v.y, 0.f);
    v.z = fmaxf(v.z, 0.f); v.w = fmaxf(v.w, 0.f);
    p[i] = v;
  }
}

extern "C" void kernel_launch(void* const* d_in, const int* in_sizes, int n_in,
                              void* d_out, int out_size, void* d_ws, size_t ws_size,
                              hipStream_t stream) {
  const float* inputs   = (const float*)d_in[0];
  const float* weight   = (const float*)d_in[1];
  const float* adj_vals = (const float*)d_in[2];
  const int*   adj_rows = (const int*)d_in[3];
  const int*   adj_cols = (const int*)d_in[4];
  float* out = (float*)d_out;
  float* H   = (float*)d_ws;                 // h = inputs @ weight, N*64 f32

  const int N = in_sizes[0] / D_IN;
  const int E = in_sizes[2];
  const int total = N * D_OUT;
  const int t4 = total / 4;                  // D_OUT=64 -> divisible by 4

  zero4_f32<<<(t4 + 255) / 256, 256, 0, stream>>>((float4*)out, t4);

  int strips  = (N + 31) / 32;               // 32 rows per wave
  int gblocks = (strips + 7) / 8;            // 8 waves per 256-thread block
  gemm_bf16x3<<<gblocks, 256, 0, stream>>>(inputs, weight, H, N);

  long long sthreads = (long long)E * 32;
  int sblocks = (int)((sthreads + 255) / 256);
  spmm_scatter<<<sblocks, 256, 0, stream>>>(H, adj_vals, adj_rows, adj_cols, out, E);

  relu4_f32<<<(t4 + 255) / 256, 256, 0, stream>>>((float4*)out, t4);
}
